// fastAPPNPTransformerBlock_55061480735302
// MI455X (gfx1250) — matrix operand order repaired and checked by reading, not compiled, not run on hardware
//
#include <hip/hip_runtime.h>
#include <cstdint>

#define MFEAT_ 266
#define MP_    288
#define ALPHA_ 0.1f
#define EPS_RF 1e-4f

typedef _Float16 h16;
typedef __attribute__((ext_vector_type(16))) _Float16 v16h;
typedef __attribute__((ext_vector_type(8)))  _Float16 v8h;
typedef __attribute__((ext_vector_type(4)))  _Float16 v4h;
typedef __attribute__((ext_vector_type(8)))  float    v8f;

// ---- WMMA fragment helpers (wave32, 16x16x32 f16 -> f32) -------------------
// A (16xK row-major slice): lane holds row M=lane&15; K halves at +0..7 / +16..23
// lda must be a multiple of 8 halves and base 16B aligned -> ds_load_b128 x2
__device__ __forceinline__ v16h load_a_frag(const h16* A, int lda) {
  int lane = threadIdx.x & 31;
  const h16* p = A + (lane & 15) * lda + ((lane >> 4) << 3);
  v8h lo = *(const v8h*)(p);
  v8h hi = *(const v8h*)(p + 16);
  return __builtin_shufflevector(lo, hi, 0, 1, 2, 3, 4, 5, 6, 7,
                                 8, 9, 10, 11, 12, 13, 14, 15);
}
// B stored N-major / K-contiguous: Bt[col][k]. lane holds col=lane&15,
// K halves at (lane>>4)*8 + 0..7 / +16..23 -> two 16B loads.
__device__ __forceinline__ v16h load_b_frag_t(const h16* B, int ldb) {
  int lane = threadIdx.x & 31;
  const h16* p = B + (lane & 15) * ldb + ((lane >> 4) << 3);
  v8h lo = *(const v8h*)(p);
  v8h hi = *(const v8h*)(p + 16);
  return __builtin_shufflevector(lo, hi, 0, 1, 2, 3, 4, 5, 6, 7,
                                 8, 9, 10, 11, 12, 13, 14, 15);
}
__device__ __forceinline__ v8f wmma32(v16h a, v16h b, v8f c) {
  return __builtin_amdgcn_wmma_f32_16x16x32_f16(false, a, false, b, (short)0, c, false, false);
}
__device__ __forceinline__ float eluf(float x) { return x > 0.f ? x : (__expf(x) - 1.f); }
__device__ __forceinline__ unsigned f2ord(float f) {
  unsigned u = __float_as_uint(f);
  return (u & 0x80000000u) ? ~u : (u | 0x80000000u);
}
__device__ __forceinline__ float ord2f(unsigned u) {
  return (u & 0x80000000u) ? __uint_as_float(u & 0x7fffffffu) : __uint_as_float(~u);
}
__device__ __forceinline__ v4h cvt4(float4 f) {
  v4h o = {(h16)f.x, (h16)f.y, (h16)f.z, (h16)f.w};
  return o;
}

// ---- utility fills ---------------------------------------------------------
__global__ void k_fill(float* p, float v, long long n) {
  long long i = (long long)blockIdx.x * 256 + threadIdx.x;
  if (i < n) p[i] = v;
}
// projht[m][k] = dn * proj[m][k] (f16, rows m>=266 zeroed). Same layout as proj.
__global__ void k_prep_proj(const float* __restrict__ proj, h16* __restrict__ projht) {
  int i = blockIdx.x * 256 + threadIdx.x;
  if (i >= MP_ * 64) return;
  int m = i / 64;
  const float dn = 0.3535533905932738f;  // 64^-0.25
  projht[i] = (h16)((m < MFEAT_) ? dn * proj[i] : 0.f);
}

// ---- fused QKV + lin1 GEMM: out[N,1024] = X[N,256] @ Wcat^T + b ------------
__global__ __launch_bounds__(256) void k_gemm_qkvg(
    const float* __restrict__ x,
    const float* __restrict__ qw, const float* __restrict__ qb,
    const float* __restrict__ kw, const float* __restrict__ kb,
    const float* __restrict__ vw, const float* __restrict__ vb,
    const float* __restrict__ l1w, const float* __restrict__ l1b,
    h16* __restrict__ qh, h16* __restrict__ kh, h16* __restrict__ vht,
    h16* __restrict__ cat, int N) {
  const int AP = 40, BP = 40;
  __shared__ __align__(16) h16 Al[128 * 40];
  __shared__ __align__(16) h16 Bt[64 * 40];   // [col][k]
  int tid = threadIdx.x, w = tid >> 5;
  int rb = blockIdx.x * 128;
  int seg = blockIdx.y >> 2;                  // 0=q 1=k 2=v 3=lin1
  int sc_base = (blockIdx.y * 64) & 255;
  const float* W = seg == 0 ? qw : seg == 1 ? kw : seg == 2 ? vw : l1w;
  const float* bias = seg == 0 ? qb : seg == 1 ? kb : seg == 2 ? vb : l1b;
  int qr = w & 3, qc = w >> 2;
  v8f c[2][2];
  for (int a = 0; a < 2; ++a) for (int b = 0; b < 2; ++b) for (int i = 0; i < 8; ++i) c[a][b][i] = 0.f;
  for (int ks = 0; ks < 8; ++ks) {
    __syncthreads();
    for (int t = 0; t < 4; ++t) {             // A tile 128x32, float4 -> v4h
      int i = tid + t * 256;
      int r = i >> 3, g = i & 7;
      v4h o = {(h16)0.f, (h16)0.f, (h16)0.f, (h16)0.f};
      if (rb + r < N) o = cvt4(*(const float4*)&x[(size_t)(rb + r) * 256 + ks * 32 + g * 4]);
      *(v4h*)&Al[r * AP + g * 4] = o;
    }
    for (int t = 0; t < 2; ++t) {             // Bt[c][k] = W[c][k]  (K-contig copy)
      int i = tid + t * 256;
      int cx = i >> 3, g = i & 7;
      *(v4h*)&Bt[cx * BP + g * 4] =
          cvt4(*(const float4*)&W[(size_t)(sc_base + cx) * 256 + ks * 32 + g * 4]);
    }
    __syncthreads();
#pragma unroll
    for (int mt = 0; mt < 2; ++mt) {
      v16h a = load_a_frag(&Al[(qr * 32 + mt * 16) * AP], AP);
#pragma unroll
      for (int nt = 0; nt < 2; ++nt) {
        v16h b = load_b_frag_t(&Bt[(qc * 32 + nt * 16) * BP], BP);
        c[mt][nt] = wmma32(a, b, c[mt][nt]);
      }
    }
  }
  int lane = tid & 31, rl0 = (lane >> 4) << 3, cl = lane & 15;
  for (int mt = 0; mt < 2; ++mt)
    for (int nt = 0; nt < 2; ++nt)
      for (int i = 0; i < 8; ++i) {
        int grow = rb + qr * 32 + mt * 16 + rl0 + i;
        if (grow >= N) continue;
        int gcl = qc * 32 + nt * 16 + cl;     // 0..63
        int sc = sc_base + gcl;
        float v = c[mt][nt][i] + bias[sc];
        int head = sc_base >> 6;
        if (seg == 0)      qh[((size_t)head * N + grow) * 64 + gcl] = (h16)v;
        else if (seg == 1) kh[((size_t)head * N + grow) * 64 + gcl] = (h16)v;
        else if (seg == 2) vht[((size_t)head * 64 + gcl) * N + grow] = (h16)v;  // V^T layout
        else               cat[(size_t)grow * 512 + sc] = (h16)eluf(v);
      }
}

// ---- performer feature map: dash = data @ (dn*proj)^T ----------------------
// IS_Q: outb = qp row-major [4][N][MP]. else: outb = dash^T, m-major [4][MP][N]
template <bool IS_Q>
__global__ __launch_bounds__(256) void k_perf(
    const h16* __restrict__ dat, const h16* __restrict__ projht,
    h16* __restrict__ outb, float* __restrict__ diagk,
    unsigned* __restrict__ kmaxb, int N) {
  const int DP = MP_ + 1;
  __shared__ float dashL[32 * (MP_ + 1)];
  __shared__ __align__(16) h16 Al[32 * 64];
  __shared__ float dg[32], mxr[32];
  __shared__ float red[256];
  int tid = threadIdx.x, w = tid >> 5;
  int h = blockIdx.y, n0 = blockIdx.x * 32;
  const h16* arow = dat + ((size_t)h * N + n0) * 64;
  *(v8h*)&Al[tid * 8] = *(const v8h*)&arow[tid * 8];   // 2048 halves, 1 v8h/thread
  __syncthreads();
  for (int nt = w; nt < 18; nt += 8) {
    for (int mt = 0; mt < 2; ++mt) {
      v8f c; for (int i = 0; i < 8; ++i) c[i] = 0.f;
#pragma unroll
      for (int s = 0; s < 2; ++s) {
        v16h a = load_a_frag(&Al[(mt * 16) * 64 + s * 32], 64);
        v16h b = load_b_frag_t(&projht[(nt * 16) * 64 + s * 32], 64);
        c = wmma32(a, b, c);
      }
      int lane = tid & 31, rl0 = (lane >> 4) << 3, cl = lane & 15;
      for (int i = 0; i < 8; ++i)
        dashL[(mt * 16 + rl0 + i) * DP + nt * 16 + cl] = c[i];
    }
  }
  __syncthreads();
  if (tid < 32) {
    float ss = 0.f;
    for (int k = 0; k < 64; ++k) { float q = (float)Al[tid * 64 + k]; ss += q * q; }
    const float dn2 = 0.3535533905932738f * 0.3535533905932738f;
    float dgv = 0.5f * dn2 * ss;
    dg[tid] = dgv;
    if (IS_Q) {
      float m = -1e30f;
      for (int j = 0; j < MFEAT_; ++j) m = fmaxf(m, dashL[tid * DP + j]);
      mxr[tid] = m;
    } else {
      diagk[(size_t)h * N + n0 + tid] = dgv;
    }
  }
  if (!IS_Q) {
    float m = -1e30f;
    for (int i = tid; i < 32 * MFEAT_; i += 256) {
      int r = i / MFEAT_, j = i % MFEAT_;
      m = fmaxf(m, dashL[r * DP + j]);
    }
    red[tid] = m;
    __syncthreads();
    for (int s = 128; s > 0; s >>= 1) {
      if (tid < s) red[tid] = fmaxf(red[tid], red[tid + s]);
      __syncthreads();
    }
    if (tid == 0) atomicMax(&kmaxb[h], f2ord(red[0]));
  }
  __syncthreads();
  const float ratio = 0.06131393f;  // 266^-0.5
  for (int i = tid; i < 32 * MP_; i += 256) {
    if (IS_Q) {
      int r = i / MP_, j = i % MP_;
      float o = (j < MFEAT_) ? ratio * (__expf(dashL[r * DP + j] - dg[r] - mxr[r]) + EPS_RF) : 0.f;
      outb[((size_t)h * N + n0 + r) * MP_ + j] = (h16)o;
    } else {
      int j = i >> 5, r = i & 31;  // coalesced along n
      float o = (j < MFEAT_) ? dashL[r * DP + j] : 0.f;
      outb[((size_t)h * MP_ + j) * N + n0 + r] = (h16)o;
    }
  }
}

// kp_t (m-major) finalize: kp = ratio*(exp(dash - diag - kmax)+eps)
__global__ void k_kp_fin(h16* kp, const float* __restrict__ diagk,
                         const unsigned* __restrict__ kmaxb, int N) {
  long long idx = (long long)blockIdx.x * 256 + threadIdx.x;
  long long tot = 4LL * MP_ * N;
  if (idx >= tot) return;
  long long n = idx % N;
  long long hm = idx / N;
  int m = (int)(hm % MP_);
  int h = (int)(hm / MP_);
  float o = 0.f;
  if (m < MFEAT_) {
    float mx = ord2f(kmaxb[h]);
    o = 0.06131393f * (__expf((float)kp[idx] - diagk[(size_t)h * N + n] - mx) + EPS_RF);
  }
  kp[idx] = (h16)o;
}

// ---- context^T[h][64][MP] = V^T @ kp ; ksum[h][MP] = colsum(kp) ------------
// kp_t: [4][MP][N] (K=n contiguous), vht: [4][64][N]
__global__ __launch_bounds__(256) void k_context(
    const h16* __restrict__ kpt, const h16* __restrict__ vht,
    float* __restrict__ ctxT, float* __restrict__ ksum, int N) {
  const int KPP = 40, VTP = 40;
  __shared__ __align__(16) h16 kpLt[288 * 40];  // [m][n-chunk]
  __shared__ __align__(16) h16 vTL[64 * 40];    // [d][n-chunk]
  int tid = threadIdx.x, w = tid >> 5;
  int h = blockIdx.y;
  int c0 = blockIdx.x * 1024;
  int rows = min(1024, N - c0);
  int nsteps = rows >> 5;
  v8f acc[3][4];
  for (int a = 0; a < 3; ++a) for (int b = 0; b < 4; ++b) for (int i = 0; i < 8; ++i) acc[a][b][i] = 0.f;
  float ks1 = 0.f, ks2 = 0.f;
  for (int s = 0; s < nsteps; ++s) {
    int nb = c0 + s * 32;
    __syncthreads();
    for (int t = 0; t < 5; ++t) {               // kp tile: 288 x 32 halves, v8h copies
      int i = tid + t * 256;
      if (i < 288 * 4) {
        int m = i >> 2, part = i & 3;
        *(v8h*)&kpLt[m * KPP + part * 8] =
            *(const v8h*)&kpt[((size_t)h * MP_ + m) * N + nb + part * 8];
      }
    }
    {                                           // v^T tile: 64 x 32 halves
      int d = tid >> 2, part = tid & 3;
      *(v8h*)&vTL[d * VTP + part * 8] =
          *(const v8h*)&vht[((size_t)h * 64 + d) * N + nb + part * 8];
    }
    __syncthreads();
    v16h a[4];
#pragma unroll
    for (int mt = 0; mt < 4; ++mt) a[mt] = load_a_frag(&vTL[(mt * 16) * VTP], VTP);
    int ti = 0;
    for (int nt = w; nt < 18; nt += 8, ++ti) {
      v16h b = load_b_frag_t(&kpLt[(nt * 16) * KPP], KPP);
#pragma unroll
      for (int mt = 0; mt < 4; ++mt) acc[ti][mt] = wmma32(a[mt], b, acc[ti][mt]);
    }
    // ksum partials: row m of kpLt is n-contiguous
    for (int g = 0; g < 4; ++g) {
      v8h vv = *(const v8h*)&kpLt[tid * KPP + g * 8];
      for (int j = 0; j < 8; ++j) ks1 += (float)vv[j];
    }
    if (tid < 32)
      for (int g = 0; g < 4; ++g) {
        v8h vv = *(const v8h*)&kpLt[(256 + tid) * KPP + g * 8];
        for (int j = 0; j < 8; ++j) ks2 += (float)vv[j];
      }
  }
  int lane = tid & 31, rl0 = (lane >> 4) << 3, cl = lane & 15;
  int ti = 0;
  for (int nt = w; nt < 18; nt += 8, ++ti)
    for (int mt = 0; mt < 4; ++mt)
      for (int i = 0; i < 8; ++i) {
        int d = mt * 16 + rl0 + i;
        int m = nt * 16 + cl;
        atomicAdd(&ctxT[((size_t)h * 64 + d) * MP_ + m], acc[ti][mt][i]);
      }
  atomicAdd(&ksum[h * MP_ + tid], ks1);
  if (tid < 32) atomicAdd(&ksum[h * MP_ + 256 + tid], ks2);
}

// ---- attention out: (qp @ context) * d_inv -> elu -> cat[:,256+h*64+d] -----
__global__ __launch_bounds__(256) void k_attn_out(
    const h16* __restrict__ qp, const float* __restrict__ ctxT,
    const float* __restrict__ ksum, h16* __restrict__ cat, int N) {
  __shared__ __align__(16) h16 cBt[64 * MP_];   // [d][m]  (B col-major: K=m contig)
  __shared__ __align__(16) h16 qA[32 * MP_];    // [n][m]
  __shared__ float ksL[MP_];
  __shared__ float dinv[32];
  int tid = threadIdx.x, w = tid >> 5;
  int h = blockIdx.y, n0 = blockIdx.x * 32;
  for (int t = 0; t < 18; ++t) {                // 64*288/4 = 4608 float4 groups
    int i = tid + t * 256;
    int d = i / 72, g = i % 72;
    *(v4h*)&cBt[d * MP_ + g * 4] = cvt4(*(const float4*)&ctxT[((size_t)h * 64 + d) * MP_ + g * 4]);
  }
  for (int t = 0; t < 5; ++t) {                 // 32*288/8 = 1152 v8h chunks
    int i = tid + t * 256;
    if (i < 1152) *(v8h*)&qA[i * 8] = *(const v8h*)&qp[((size_t)h * N + n0) * MP_ + i * 8];
  }
  for (int i = tid; i < MP_; i += 256) ksL[i] = ksum[h * MP_ + i];
  __syncthreads();
  if (tid < 32) {
    float d = 0.f;
    for (int j = 0; j < MP_; ++j) d += (float)qA[tid * MP_ + j] * ksL[j];
    dinv[tid] = 1.f / d;
  }
  __syncthreads();
  int mt = w & 1, nt = w >> 1;
  v8f c; for (int i = 0; i < 8; ++i) c[i] = 0.f;
  for (int s = 0; s < 9; ++s) {
    v16h a = load_a_frag(&qA[(mt * 16) * MP_ + s * 32], MP_);
    v16h b = load_b_frag_t(&cBt[(nt * 16) * MP_ + s * 32], MP_);
    c = wmma32(a, b, c);
  }
  int lane = tid & 31, rl0 = (lane >> 4) << 3, cl = lane & 15;
  for (int i = 0; i < 8; ++i) {
    int r = mt * 16 + rl0 + i;
    int d = nt * 16 + cl;
    float v = c[i] * dinv[r];
    cat[(size_t)(n0 + r) * 512 + 256 + h * 64 + d] = (h16)eluf(v);
  }
}

// ---- lin2: gnn2[N,32] = elu(cat[N,512] @ l2w^T + b) ------------------------
__global__ __launch_bounds__(256) void k_lin2(
    const h16* __restrict__ cat, const float* __restrict__ l2w,
    const float* __restrict__ l2b, float* __restrict__ gnn2,
    float* __restrict__ x0, int N) {
  const int AP = 40;
  __shared__ __align__(16) h16 Bt[32 * 512];    // [col][k] = l2w rows (K contig)
  __shared__ __align__(16) h16 Al[64 * 40];
  int tid = threadIdx.x, w = tid >> 5;
  int rb = blockIdx.x * 64;
  for (int t = 0; t < 16; ++t) {                // 32*512/4 = 4096 float4 groups
    int i = tid + t * 256;
    int cx = i >> 7, g = i & 127;
    *(v4h*)&Bt[cx * 512 + g * 4] = cvt4(*(const float4*)&l2w[(size_t)cx * 512 + g * 4]);
  }
  int mt = w & 3, nt = w >> 2;
  v8f c; for (int i = 0; i < 8; ++i) c[i] = 0.f;
  for (int s = 0; s < 16; ++s) {
    __syncthreads();
    {
      int r = tid >> 2, g = tid & 3;            // 64x32 halves, v8h copies
      v8h o = {(h16)0.f, (h16)0.f, (h16)0.f, (h16)0.f, (h16)0.f, (h16)0.f, (h16)0.f, (h16)0.f};
      if (rb + r < N) o = *(const v8h*)&cat[(size_t)(rb + r) * 512 + s * 32 + g * 8];
      *(v8h*)&Al[r * AP + g * 8] = o;
    }
    __syncthreads();
    v16h a = load_a_frag(&Al[(mt * 16) * AP], AP);
    v16h b = load_b_frag_t(&Bt[(nt * 16) * 512 + s * 32], 512);
    c = wmma32(a, b, c);
  }
  int lane = tid & 31, rl0 = (lane >> 4) << 3, cl = lane & 15;
  for (int i = 0; i < 8; ++i) {
    int r = rb + mt * 16 + rl0 + i;
    int col = nt * 16 + cl;
    if (r < N) {
      float v = eluf(c[i] + l2b[col]);
      gnn2[(size_t)r * 32 + col] = v;
      x0[(size_t)r * 32 + col] = v;
    }
  }
}

// ---- APPNP -----------------------------------------------------------------
__global__ void k_deg_edges(float* deg, const long long* __restrict__ ei, long long E) {
  long long e = (long long)blockIdx.x * 256 + threadIdx.x;
  if (e < E) atomicAdd(&deg[ei[E + e]], 1.f);
}
__global__ void k_norm(const float* __restrict__ deg, float* dis, float* selfw, int N) {
  int i = blockIdx.x * 256 + threadIdx.x;
  if (i < N) { float d = rsqrtf(deg[i]); dis[i] = d; selfw[i] = d * d; }
}
__global__ void k_ew(const float* __restrict__ dis, const long long* __restrict__ ei,
                     float* ew, long long E) {
  long long e = (long long)blockIdx.x * 256 + threadIdx.x;
  if (e < E) ew[e] = dis[ei[e]] * dis[ei[E + e]];
}
__global__ void k_prop_init(float* __restrict__ xn, const float* __restrict__ h0,
                            const float* __restrict__ selfw, const float* __restrict__ xc,
                            int N) {
  long long idx = (long long)blockIdx.x * 256 + threadIdx.x;
  if (idx < (long long)N * 32) {
    int i = (int)(idx >> 5);
    xn[idx] = ALPHA_ * h0[idx] + (1.f - ALPHA_) * selfw[i] * xc[idx];
  }
}
__global__ void k_prop_edges(float* __restrict__ xn, const float* __restrict__ xc,
                             const long long* __restrict__ ei, const float* __restrict__ ew,
                             long long E) {
  long long gid = (long long)blockIdx.x * 256 + threadIdx.x;
  if (gid >= E * 8) return;
  long long e = gid >> 3;
  int part = (int)(gid & 7);
  long long s = ei[e], d = ei[E + e];
  float w = (1.f - ALPHA_) * ew[e];
  const float4 v = *(const float4*)&xc[s * 32 + part * 4];
  float* dst = &xn[d * 32 + part * 4];
  atomicAdd(dst + 0, w * v.x);
  atomicAdd(dst + 1, w * v.y);
  atomicAdd(dst + 2, w * v.z);
  atomicAdd(dst + 3, w * v.w);
}
__global__ void k_logsoftmax(const float* __restrict__ xf, float* __restrict__ out, int N) {
  int n = blockIdx.x * 256 + threadIdx.x;
  if (n >= N) return;
  float v[32], m = -1e30f;
  for (int c = 0; c < 32; ++c) { v[c] = xf[(size_t)n * 32 + c]; m = fmaxf(m, v[c]); }
  float s = 0.f;
  for (int c = 0; c < 32; ++c) s += __expf(v[c] - m);
  float ls = __logf(s);
  for (int c = 0; c < 32; ++c) {
    out[(size_t)n * 32 + c] = v[c] - m - ls;
    out[(size_t)N * 32 + (size_t)n * 32 + c] = v[c];
  }
}

// ---- host ------------------------------------------------------------------
extern "C" void kernel_launch(void* const* d_in, const int* in_sizes, int n_in,
                              void* d_out, int out_size, void* d_ws, size_t ws_size,
                              hipStream_t stream) {
  const float* x   = (const float*)d_in[0];
  const long long* ei = (const long long*)d_in[1];
  const float* l1w = (const float*)d_in[2];
  const float* l1b = (const float*)d_in[3];
  const float* l2w = (const float*)d_in[4];
  const float* l2b = (const float*)d_in[5];
  const float* qw  = (const float*)d_in[6];
  const float* qb  = (const float*)d_in[7];
  const float* kw  = (const float*)d_in[8];
  const float* kb  = (const float*)d_in[9];
  const float* vw  = (const float*)d_in[10];
  const float* vb  = (const float*)d_in[11];
  const float* proj = (const float*)d_in[12];
  const long long N = in_sizes[0] / 256;
  const long long E = in_sizes[1] / 2;

  char* ws = (char*)d_ws;
  size_t cur = 0;
  auto alloc = [&](size_t bytes) {
    size_t o = cur; cur += (bytes + 255) & ~(size_t)255; return o;
  };
  h16*   qh    = (h16*)(ws + alloc((size_t)4 * N * 64 * 2));
  h16*   kh    = (h16*)(ws + alloc((size_t)4 * N * 64 * 2));
  h16*   vht   = (h16*)(ws + alloc((size_t)4 * N * 64 * 2));  // [4][64][N]
  h16*   cat   = (h16*)(ws + alloc((size_t)N * 512 * 2));
  h16*   qp    = (h16*)(ws + alloc((size_t)4 * N * MP_ * 2)); // [4][N][MP]
  h16*   kpt   = (h16*)(ws + alloc((size_t)4 * N * MP_ * 2)); // [4][MP][N]
  h16*   projht = (h16*)(ws + alloc((size_t)MP_ * 64 * 2));
  float* diagk = (float*)(ws + alloc((size_t)4 * N * 4));
  unsigned* kmaxb = (unsigned*)(ws + alloc(256));
  float* ctxT  = (float*)(ws + alloc((size_t)4 * 64 * MP_ * 4));
  float* ksum  = (float*)(ws + alloc((size_t)4 * MP_ * 4));
  float* gnn2  = (float*)(ws + alloc((size_t)N * 32 * 4));
  float* x0    = (float*)(ws + alloc((size_t)N * 32 * 4));
  float* x1    = (float*)(ws + alloc((size_t)N * 32 * 4));
  float* deg   = (float*)(ws + alloc((size_t)N * 4));
  float* dis   = (float*)(ws + alloc((size_t)N * 4));
  float* selfw = (float*)(ws + alloc((size_t)N * 4));
  float* ew    = (float*)(ws + alloc((size_t)E * 4));
  (void)n_in; (void)out_size; (void)ws_size;

  auto gb = [](long long n) { return (int)((n + 255) / 256); };

  k_fill<<<gb(4 * 64 * MP_), 256, 0, stream>>>(ctxT, 0.f, 4LL * 64 * MP_);
  k_fill<<<gb(4 * MP_), 256, 0, stream>>>(ksum, 0.f, 4LL * MP_);
  k_fill<<<1, 256, 0, stream>>>((float*)kmaxb, 0.f, 4);
  k_fill<<<gb(N), 256, 0, stream>>>(deg, 1.f, N);  // self loop
  k_prep_proj<<<gb(MP_ * 64), 256, 0, stream>>>(proj, projht);

  dim3 gq((unsigned)((N + 127) / 128), 16);
  k_gemm_qkvg<<<gq, 256, 0, stream>>>(x, qw, qb, kw, kb, vw, vb, l1w, l1b,
                                      qh, kh, vht, cat, (int)N);
  dim3 gp((unsigned)(N / 32), 4);
  k_perf<true><<<gp, 256, 0, stream>>>(qh, projht, qp, nullptr, nullptr, (int)N);
  k_perf<false><<<gp, 256, 0, stream>>>(kh, projht, kpt, diagk, kmaxb, (int)N);
  k_kp_fin<<<gb(4LL * N * MP_), 256, 0, stream>>>(kpt, diagk, kmaxb, (int)N);
  dim3 gc((unsigned)((N + 1023) / 1024), 4);
  k_context<<<gc, 256, 0, stream>>>(kpt, vht, ctxT, ksum, (int)N);
  k_attn_out<<<gp, 256, 0, stream>>>(qp, ctxT, ksum, cat, (int)N);
  k_lin2<<<(unsigned)((N + 63) / 64), 256, 0, stream>>>(cat, l2w, l2b, gnn2, x0, (int)N);

  k_deg_edges<<<gb(E), 256, 0, stream>>>(deg, ei, E);
  k_norm<<<gb(N), 256, 0, stream>>>(deg, dis, selfw, (int)N);
  k_ew<<<gb(E), 256, 0, stream>>>(dis, ei, ew, E);
  float* xc = x0; float* xn = x1;
  for (int t = 0; t < 10; ++t) {
    k_prop_init<<<gb(N * 32), 256, 0, stream>>>(xn, gnn2, selfw, xc, (int)N);
    k_prop_edges<<<gb(E * 8), 256, 0, stream>>>(xn, xc, ei, ew, E);
    float* tmp = xc; xc = xn; xn = tmp;
  }
  k_logsoftmax<<<gb(N), 256, 0, stream>>>(xc, (float*)d_out, (int)N);
}
